// GraphEmbedder_57329223467060
// MI455X (gfx1250) — compile-verified, matrix operand order checked by reading, not run on hardware
//
#include <hip/hip_runtime.h>
#include <hip/hip_fp16.h>

#define SLOPE 0.2f

typedef __attribute__((ext_vector_type(16))) _Float16 v16h;
typedef __attribute__((ext_vector_type(8)))  _Float16 v8h;
typedef __attribute__((ext_vector_type(8)))  float    v8f;

__device__ __forceinline__ float leaky(float v) { return v > 0.f ? v : SLOPE * v; }

// float atomic max via sign-split int/uint atomics (valid with -inf init)
__device__ __forceinline__ void atomicMaxF32(float* addr, float val) {
    if (val >= 0.f) atomicMax((int*)addr, __float_as_int(val));
    else            atomicMin((unsigned int*)addr, __float_as_uint(val));
}

// ---------------- init / elementwise ----------------
__global__ void k_fill(float* p, float v, int n) {
    int i = blockIdx.x * blockDim.x + threadIdx.x;
    if (i < n) p[i] = v;
}

__global__ void k_preprocess_x(const float* __restrict__ x, float* __restrict__ xp, int N) {
    int i = blockIdx.x * blockDim.x + threadIdx.x;
    if (i >= N) return;
    float r[6];
#pragma unroll
    for (int d = 0; d < 6; ++d) r[d] = x[i * 6 + d];
    if (r[2] == 1.0f) r[1] = (r[1] - r[0]) * 0.01f;
#pragma unroll
    for (int d = 0; d < 6; ++d) xp[i * 6 + d] = r[d];
}

// column sums of preprocessed edge_attr (mean = sum * invE in consumer)
__global__ void k_edge_mean(const float* __restrict__ ea, float* __restrict__ sum, int E) {
    int i = blockIdx.x * blockDim.x + threadIdx.x;
    int stride = gridDim.x * blockDim.x;
    float acc[11];
#pragma unroll
    for (int d = 0; d < 11; ++d) acc[d] = 0.f;
    for (int e = i; e < E; e += stride) {
        float fix = (ea[e * 11 + 9] == 1.0f) ? (ea[e * 11 + 1] - ea[e * 11 + 0]) * 0.01f : ea[e * 11 + 1];
#pragma unroll
        for (int d = 0; d < 11; ++d) acc[d] += (d == 1) ? fix : ea[e * 11 + d];
    }
#pragma unroll
    for (int d = 0; d < 11; ++d) atomicAdd(&sum[d], acc[d]);
}

// ---------------- small-K GEMM (layer 0, K=6): out[M,256] = A[M,K]@W[K,256]+b ----------------
__global__ void k_gemm_small(const float* __restrict__ A, int K, const float* __restrict__ W,
                             const float* __restrict__ b, float* __restrict__ out, int M) {
    int t = blockIdx.x * blockDim.x + threadIdx.x;
    if (t >= M * 256) return;
    int m = t >> 8, n = t & 255;
    float acc = b[n];
    for (int k = 0; k < K; ++k) acc = fmaf(A[m * K + k], W[k * 256 + n], acc);
    out[t] = acc;
}

// ---------------- layer-0 logits (edge features fused, never materialized) ----------------
__global__ void k_logits0(const float* __restrict__ xl, const float* __restrict__ xr,
                          const int* __restrict__ ei, const float* __restrict__ ea,
                          const float* __restrict__ esum, float invE,
                          const float* __restrict__ We, const float* __restrict__ att,
                          float* __restrict__ logits, float* __restrict__ mx,
                          int E, int Etot) {
    int t = blockIdx.x * blockDim.x + threadIdx.x;
    if (t >= Etot * 8) return;
    int e = t >> 3, h = t & 7;
    int src, tgt;
    float ev[11];
    if (e < E) {
        src = ei[e]; tgt = ei[E + e];
#pragma unroll
        for (int d = 0; d < 11; ++d) ev[d] = ea[e * 11 + d];
        if (ev[9] == 1.0f) ev[1] = (ev[1] - ev[0]) * 0.01f;
    } else {
        src = tgt = e - E;
#pragma unroll
        for (int d = 0; d < 11; ++d) ev[d] = esum[d] * invE;
    }
    float logit = 0.f;
    int base = h * 32;
    for (int c = 0; c < 32; ++c) {
        int hc = base + c;
        float eec = 0.f;
#pragma unroll
        for (int d = 0; d < 11; ++d) eec = fmaf(ev[d], We[d * 256 + hc], eec);
        float m = xl[src * 256 + hc] + xr[tgt * 256 + hc] + eec;
        logit = fmaf(att[hc], leaky(m), logit);
    }
    logits[t] = logit;
    atomicMaxF32(&mx[tgt * 8 + h], logit);
}

// ---------------- layer-1 logits (no edge features) ----------------
__global__ void k_logits1(const float* __restrict__ xl, const float* __restrict__ xr,
                          const int* __restrict__ ei, const float* __restrict__ att,
                          float* __restrict__ logits, float* __restrict__ mx,
                          int E, int Etot) {
    int t = blockIdx.x * blockDim.x + threadIdx.x;
    if (t >= Etot * 8) return;
    int e = t >> 3, h = t & 7;
    int src, tgt;
    if (e < E) { src = ei[e]; tgt = ei[E + e]; } else { src = tgt = e - E; }
    float logit = 0.f;
    int base = h * 32;
    for (int c = 0; c < 32; ++c) {
        int hc = base + c;
        float m = xl[src * 256 + hc] + xr[tgt * 256 + hc];
        logit = fmaf(att[hc], leaky(m), logit);
    }
    logits[t] = logit;
    atomicMaxF32(&mx[tgt * 8 + h], logit);
}

// exp(logit - max) in place; accumulate denominator
__global__ void k_expden(float* __restrict__ logits, const float* __restrict__ mx,
                         float* __restrict__ den, const int* __restrict__ ei,
                         int E, int Etot) {
    int t = blockIdx.x * blockDim.x + threadIdx.x;
    if (t >= Etot * 8) return;
    int e = t >> 3, h = t & 7;
    int tgt = (e < E) ? ei[E + e] : e - E;
    float ex = expf(logits[t] - mx[tgt * 8 + h]);
    logits[t] = ex;
    atomicAdd(&den[tgt * 8 + h], ex);
}

// out[tgt] += alpha * xl[src]   (out pre-initialized with 0 or bias)
__global__ void k_aggregate(const float* __restrict__ ex, const float* __restrict__ den,
                            const float* __restrict__ xl, const int* __restrict__ ei,
                            float* __restrict__ out, int E, int Etot) {
    int t = blockIdx.x * blockDim.x + threadIdx.x;
    if (t >= Etot * 8) return;
    int e = t >> 3, h = t & 7;
    int src, tgt;
    if (e < E) { src = ei[e]; tgt = ei[E + e]; } else { src = tgt = e - E; }
    float alpha = ex[t] / den[tgt * 8 + h];
    int ob = tgt * 256 + h * 32, ib = src * 256 + h * 32;
#pragma unroll
    for (int c = 0; c < 32; ++c) atomicAdd(&out[ob + c], alpha * xl[ib + c]);
}

__global__ void k_bias_elu(float* __restrict__ hbuf, const float* __restrict__ bias, int total) {
    int t = blockIdx.x * blockDim.x + threadIdx.x;
    if (t >= total) return;
    float v = hbuf[t] + bias[t & 255];
    hbuf[t] = v > 0.f ? v : (expf(v) - 1.0f);
}

__global__ void k_f32_to_f16(const float* __restrict__ a, _Float16* __restrict__ b, int n) {
    int t = blockIdx.x * blockDim.x + threadIdx.x;
    if (t < n) b[t] = (_Float16)a[t];
}

// Wt[n*256+k] = (f16)W[k*256+n]  (n-major so B fragments load contiguously in k)
__global__ void k_wtrans16(const float* __restrict__ W, _Float16* __restrict__ Wt) {
    int t = blockIdx.x * blockDim.x + threadIdx.x;
    if (t >= 65536) return;
    int n = t >> 8, k = t & 255;
    Wt[n * 256 + k] = (_Float16)W[k * 256 + n];
}

__global__ void k_init_bias(float* __restrict__ out, const float* __restrict__ bias, int total) {
    int t = blockIdx.x * blockDim.x + threadIdx.x;
    if (t < total) out[t] = bias[t & 255];
}

// ---------------- WMMA GEMM: C[M,256] = A16[M,256] @ B(Bt16 n-major)[256,256] + bias ----------------
// one wave per 16x16 output tile, K loop = 8 x v_wmma_f32_16x16x32_f16
__global__ void k_wmma_gemm(const _Float16* __restrict__ A16, const _Float16* __restrict__ Bt16,
                            const float* __restrict__ bias, float* __restrict__ Cout, int Mtiles) {
    int wid  = (blockIdx.x * blockDim.x + threadIdx.x) >> 5;
    int lane = threadIdx.x & 31;
    int tileM = wid >> 4;   // 256/16 = 16 N-tiles
    int tileN = wid & 15;
    if (tileM >= Mtiles) return;
    int m       = lane & 15;   // A row within tile / B col within tile / D col
    int halfSel = lane >> 4;   // K half selector per ISA 16-bit A/B layout
    const _Float16* arow = A16  + (size_t)(tileM * 16 + m) * 256;
    const _Float16* brow = Bt16 + (size_t)(tileN * 16 + m) * 256;
    v8f acc = {};
    union U { v16h v; v8h h[2]; };
    for (int kk = 0; kk < 256; kk += 32) {
        U a, b;
        a.h[0] = *(const v8h*)(arow + kk + halfSel * 8);
        a.h[1] = *(const v8h*)(arow + kk + 16 + halfSel * 8);
        b.h[0] = *(const v8h*)(brow + kk + halfSel * 8);
        b.h[1] = *(const v8h*)(brow + kk + 16 + halfSel * 8);
        acc = __builtin_amdgcn_wmma_f32_16x16x32_f16(false, a.v, false, b.v,
                                                     (short)0, acc, false, false);
    }
    float bv = bias[tileN * 16 + m];
#pragma unroll
    for (int r = 0; r < 8; ++r) {
        int mr = halfSel * 8 + r;   // D row per ISA 32-bit C/D layout
        Cout[(size_t)(tileM * 16 + mr) * 256 + tileN * 16 + m] = acc[r] + bv;
    }
}

extern "C" void kernel_launch(void* const* d_in, const int* in_sizes, int n_in,
                              void* d_out, int out_size, void* d_ws, size_t ws_size,
                              hipStream_t stream) {
    const float* x     = (const float*)d_in[0];
    const int*   ei    = (const int*)  d_in[1];
    const float* eattr = (const float*)d_in[2];
    const float* Wl0   = (const float*)d_in[3];
    const float* bl0   = (const float*)d_in[4];
    const float* Wr0   = (const float*)d_in[5];
    const float* br0   = (const float*)d_in[6];
    const float* We0   = (const float*)d_in[7];
    const float* att0  = (const float*)d_in[8];
    const float* bias0 = (const float*)d_in[9];
    const float* Wl1   = (const float*)d_in[10];
    const float* bl1   = (const float*)d_in[11];
    const float* Wr1   = (const float*)d_in[12];
    const float* br1   = (const float*)d_in[13];
    const float* att1  = (const float*)d_in[14];
    const float* bias1 = (const float*)d_in[15];

    const int N    = in_sizes[0] / 6;
    const int E    = in_sizes[1] / 2;
    const int Etot = E + N;

    // workspace carve-out (256B aligned)
    size_t off = 0;
    auto alloc = [&](size_t bytes) -> char* {
        off = (off + 255) & ~size_t(255);
        char* p = (char*)d_ws + off;
        off += bytes;
        return p;
    };
    float*     xl     = (float*)alloc((size_t)N * 256 * 4);
    float*     xr     = (float*)alloc((size_t)N * 256 * 4);
    float*     hbuf   = (float*)alloc((size_t)N * 256 * 4);
    float*     logits = (float*)alloc((size_t)Etot * 8 * 4);
    float*     mx     = (float*)alloc((size_t)N * 8 * 4);
    float*     den    = (float*)alloc((size_t)N * 8 * 4);
    float*     xp     = (float*)alloc((size_t)N * 6 * 4);
    float*     esum   = (float*)alloc(64);
    _Float16*  h16    = (_Float16*)alloc((size_t)N * 256 * 2);
    _Float16*  w16    = (_Float16*)alloc((size_t)256 * 256 * 2);

    const int B = 256;
    auto gr = [&](long n) { return (int)((n + B - 1) / B); };
    const int NC = N * 256;       // node-channel total
    const int EH = Etot * 8;      // edge-head total
    const int Mtiles = N / 16;    // 1250

    // ---- preprocessing ----
    k_preprocess_x<<<gr(N), B, 0, stream>>>(x, xp, N);
    k_fill<<<1, 16, 0, stream>>>(esum, 0.f, 16);
    k_edge_mean<<<64, B, 0, stream>>>(eattr, esum, E);

    // ---- layer 0 ----
    k_gemm_small<<<gr(NC), B, 0, stream>>>(xp, 6, Wl0, bl0, xl, N);
    k_gemm_small<<<gr(NC), B, 0, stream>>>(xp, 6, Wr0, br0, xr, N);
    k_fill<<<gr(N * 8), B, 0, stream>>>(mx, -INFINITY, N * 8);
    k_fill<<<gr(N * 8), B, 0, stream>>>(den, 0.f, N * 8);
    k_logits0<<<gr(EH), B, 0, stream>>>(xl, xr, ei, eattr, esum, 1.0f / (float)E,
                                        We0, att0, logits, mx, E, Etot);
    k_expden<<<gr(EH), B, 0, stream>>>(logits, mx, den, ei, E, Etot);
    k_fill<<<gr(NC), B, 0, stream>>>(hbuf, 0.f, NC);
    k_aggregate<<<gr(EH), B, 0, stream>>>(logits, den, xl, ei, hbuf, E, Etot);
    k_bias_elu<<<gr(NC), B, 0, stream>>>(hbuf, bias0, NC);

    // ---- layer 1 GEMMs via WMMA (f16 in, f32 accumulate) ----
    k_f32_to_f16<<<gr(NC), B, 0, stream>>>(hbuf, h16, NC);
    k_wtrans16<<<gr(65536), B, 0, stream>>>(Wl1, w16);
    k_wmma_gemm<<<Mtiles * 2, B, 0, stream>>>(h16, w16, bl1, xl, Mtiles);
    k_wtrans16<<<gr(65536), B, 0, stream>>>(Wr1, w16);
    k_wmma_gemm<<<Mtiles * 2, B, 0, stream>>>(h16, w16, br1, xr, Mtiles);

    // ---- layer 1 attention + aggregation into d_out ----
    k_fill<<<gr(N * 8), B, 0, stream>>>(mx, -INFINITY, N * 8);
    k_fill<<<gr(N * 8), B, 0, stream>>>(den, 0.f, N * 8);
    k_logits1<<<gr(EH), B, 0, stream>>>(xl, xr, ei, att1, logits, mx, E, Etot);
    k_expden<<<gr(EH), B, 0, stream>>>(logits, mx, den, ei, E, Etot);
    k_init_bias<<<gr(NC), B, 0, stream>>>((float*)d_out, bias1, NC);
    k_aggregate<<<gr(EH), B, 0, stream>>>(logits, den, xl, ei, (float*)d_out, E, Etot);
}
